// AttentionPatcher_21749714387544
// MI455X (gfx1250) — matrix operand order repaired and checked by reading, not compiled, not run on hardware
//
#include <hip/hip_runtime.h>

// ---------------------------------------------------------------------------
// Problem constants (from the reference)
// ---------------------------------------------------------------------------
#define H_   32
#define KV_  8
#define D_   128
#define S_   2048
#define HID_ 4096
#define G_   (H_ / KV_)   // 4

// ---------------------------------------------------------------------------
// Types for CDNA5 WMMA
// ---------------------------------------------------------------------------
typedef __attribute__((ext_vector_type(4)))  unsigned int u32x4;
typedef __attribute__((ext_vector_type(8)))  float        v8f;
typedef __attribute__((ext_vector_type(16))) __bf16       v16bf;

union FragU { u32x4 q[2]; v16bf v; };

__device__ __forceinline__ v8f wmma_bf16(v16bf a, v16bf b, v8f c) {
  // D = A(16x32 bf16) * B(32x16 bf16) + C(16x16 f32)
  return __builtin_amdgcn_wmma_f32_16x16x32_bf16(
      /*neg_a=*/false, a, /*neg_b=*/false, b,
      /*c_mod=*/(short)0, c, /*reuse_a=*/false, /*reuse_b=*/false);
}

// f32 -> bf16 with round-to-nearest-even
__device__ __forceinline__ unsigned short f2bf(float f) {
  union { float f; unsigned u; } v; v.f = f;
  unsigned r = v.u + 0x7FFFu + ((v.u >> 16) & 1u);
  return (unsigned short)(r >> 16);
}

// ---------------------------------------------------------------------------
// CDNA5 async global->LDS copy (ISA §15.18.3 GLOBAL_LOAD_ASYNC_TO_LDS_B128,
// tracked by ASYNCcnt). The LDS address is the low 32 bits of the generic
// pointer (flat LDS aperture, workgroup-relative).
// ---------------------------------------------------------------------------
__device__ __forceinline__ void async_copy16(unsigned short* lds,
                                             const unsigned short* g) {
  unsigned laddr = (unsigned)(size_t)lds;
  unsigned long long gaddr = (unsigned long long)(size_t)g;
  asm volatile("global_load_async_to_lds_b128 %0, %1, off"
               :: "v"(laddr), "v"(gaddr)
               : "memory");
}
__device__ __forceinline__ void wait_async0() {
  asm volatile("s_wait_asynccnt 0x0" ::: "memory");
}

// A-fragment: row-major [row][k] bf16 in LDS, 32 k-values starting at kbase.
// Lane layout (ISA 7.12.2): row = lane%16; e<8 -> k = kbase+8*half+e,
// e>=8 -> k = kbase+16+8*half+(e-8).
__device__ __forceinline__ v16bf load_afrag(const unsigned short* sm, int row,
                                            int st, int kbase, int half) {
  FragU u;
  const unsigned short* p = sm + row * st + kbase + 8 * half;
  u.q[0] = *(const u32x4*)(p);
  u.q[1] = *(const u32x4*)(p + 16);
  return u.v;
}

// B-fragment from B^T stored row-major [n][k]: lane col n = lane%16,
// e<8 -> k = kbase+8*half+e, e>=8 -> k = kbase+16+8*half+(e-8).
__device__ __forceinline__ v16bf load_bfrag(const unsigned short* sm, int n,
                                            int st, int kbase, int half) {
  FragU u;
  const unsigned short* p = sm + n * st + kbase + 8 * half;
  u.q[0] = *(const u32x4*)(p);
  u.q[1] = *(const u32x4*)(p + 16);
  return u.v;
}

// wave32: reduce across the 16 lanes of a half-wave (C-matrix row group)
__device__ __forceinline__ float rmax16(float v) {
  for (int m = 1; m < 16; m <<= 1) v = fmaxf(v, __shfl_xor(v, m, 32));
  return v;
}
__device__ __forceinline__ float rsum16(float v) {
  for (int m = 1; m < 16; m <<= 1) v += __shfl_xor(v, m, 32);
  return v;
}

// ---------------------------------------------------------------------------
// Elementwise f32 -> bf16 cast
// ---------------------------------------------------------------------------
__global__ void cast_bf16_kernel(const float* __restrict__ in,
                                 unsigned short* __restrict__ out, long n) {
  long i = (long)blockIdx.x * blockDim.x + threadIdx.x;
  long stride = (long)gridDim.x * blockDim.x;
  for (; i < n; i += stride) out[i] = f2bf(in[i]);
}

// ---------------------------------------------------------------------------
// x [HID][S] f32  ->  xT [S][HID] bf16   (LDS 32x33 tile transpose)
// ---------------------------------------------------------------------------
__global__ __launch_bounds__(256)
void transpose_cast_kernel(const float* __restrict__ x,
                           unsigned short* __restrict__ xT) {
  __shared__ float t[32][33];
  const int c0 = blockIdx.x * 32;  // along S
  const int r0 = blockIdx.y * 32;  // along HID
  const int tx = threadIdx.x, ty = threadIdx.y;  // 32 x 8
#pragma unroll
  for (int j = 0; j < 32; j += 8)
    t[ty + j][tx] = x[(size_t)(r0 + ty + j) * S_ + c0 + tx];
  __syncthreads();
#pragma unroll
  for (int j = 0; j < 32; j += 8)
    xT[(size_t)(c0 + ty + j) * HID_ + r0 + tx] = f2bf(t[tx][ty + j]);
}

// ---------------------------------------------------------------------------
// GEMM: C[M][N] f32 = A[M][K] bf16 * (BT[N][K] bf16)^T + bias
// Block tile 128x128x32, 256 threads = 8 waves (4 along M x 2 along N),
// each wave owns 32x64 = 2x4 WMMA accumulators. Double-buffered LDS filled
// by GLOBAL_LOAD_ASYNC_TO_LDS_B128: next tile's DMA overlaps this tile's
// WMMAs; one s_wait_asynccnt + barrier per K-step.
// ---------------------------------------------------------------------------
#define BM 128
#define BN 128
#define BKT 32
#define LDT 40  // padded LDS row stride (ushorts)

__global__ __launch_bounds__(256)
void gemm_bf16_kernel(const unsigned short* __restrict__ A,
                      const unsigned short* __restrict__ BT,
                      const float* __restrict__ bias,
                      float* __restrict__ C, int M, int N, int K) {
  __shared__ __attribute__((aligned(16))) unsigned short As[2][BM * LDT];
  __shared__ __attribute__((aligned(16))) unsigned short Bs[2][BN * LDT];

  const int tid = threadIdx.x;
  const int wave = tid >> 5, lane = tid & 31;
  const int half = lane >> 4, ln = lane & 15;
  const int wm = wave >> 1, wn = wave & 1;  // 4 x 2 wave grid
  const int m0 = blockIdx.y * BM, n0 = blockIdx.x * BN;

  // this thread's two 16B chunks per matrix per tile
  const int r0c = (tid) >> 2, c0c = ((tid) & 3) << 3;
  const int r1c = (tid + 256) >> 2, c1c = ((tid + 256) & 3) << 3;

  const v8f vzero = {0, 0, 0, 0, 0, 0, 0, 0};
  v8f acc[2][4];
#pragma unroll
  for (int i = 0; i < 2; ++i)
#pragma unroll
    for (int j = 0; j < 4; ++j) acc[i][j] = vzero;

  const int nkt = K / BKT;
  // preload tile 0 into buffer 0
  {
    async_copy16(&As[0][r0c * LDT + c0c], A + (size_t)(m0 + r0c) * K + c0c);
    async_copy16(&As[0][r1c * LDT + c1c], A + (size_t)(m0 + r1c) * K + c1c);
    async_copy16(&Bs[0][r0c * LDT + c0c], BT + (size_t)(n0 + r0c) * K + c0c);
    async_copy16(&Bs[0][r1c * LDT + c1c], BT + (size_t)(n0 + r1c) * K + c1c);
  }
  wait_async0();
  __syncthreads();

  for (int kt = 0; kt < nkt; ++kt) {
    const int cur = kt & 1;
    if (kt + 1 < nkt) {  // issue next tile's DMA before computing
      const int nxt = cur ^ 1;
      const int kn = (kt + 1) * BKT;
      async_copy16(&As[nxt][r0c * LDT + c0c],
                   A + (size_t)(m0 + r0c) * K + kn + c0c);
      async_copy16(&As[nxt][r1c * LDT + c1c],
                   A + (size_t)(m0 + r1c) * K + kn + c1c);
      async_copy16(&Bs[nxt][r0c * LDT + c0c],
                   BT + (size_t)(n0 + r0c) * K + kn + c0c);
      async_copy16(&Bs[nxt][r1c * LDT + c1c],
                   BT + (size_t)(n0 + r1c) * K + kn + c1c);
    }
    v16bf af0 = load_afrag(As[cur], wm * 32 + ln, LDT, 0, half);
    v16bf af1 = load_afrag(As[cur], wm * 32 + 16 + ln, LDT, 0, half);
#pragma unroll
    for (int nt = 0; nt < 4; ++nt) {
      v16bf bf = load_bfrag(Bs[cur], wn * 64 + nt * 16 + ln, LDT, 0, half);
      acc[0][nt] = wmma_bf16(af0, bf, acc[0][nt]);
      acc[1][nt] = wmma_bf16(af1, bf, acc[1][nt]);
    }
    wait_async0();
    __syncthreads();
  }
  // epilogue: C layout row = r + 8*half, col = lane%16
#pragma unroll
  for (int mt = 0; mt < 2; ++mt)
#pragma unroll
    for (int nt = 0; nt < 4; ++nt)
#pragma unroll
      for (int r = 0; r < 8; ++r) {
        int m = m0 + wm * 32 + mt * 16 + r + 8 * half;
        int n = n0 + wn * 64 + nt * 16 + ln;
        float v = acc[mt][nt][r];
        if (bias) v += bias[m];
        C[(size_t)m * N + n] = v;
      }
}

// ---------------------------------------------------------------------------
// RoPE + layout change: in [NH*D][S] f32 -> out [NH][S][D] bf16
// ---------------------------------------------------------------------------
__global__ void rope_kernel(const float* __restrict__ qf,
                            const float* __restrict__ cosp,
                            const float* __restrict__ sinp,
                            unsigned short* __restrict__ outb, long n) {
  long idx = (long)blockIdx.x * blockDim.x + threadIdx.x;
  if (idx >= n) return;
  int d = idx & (D_ - 1);
  long t = idx >> 7;          // /D
  int s = t & (S_ - 1);
  long h = t >> 11;           // /S
  float v = qf[(h * D_ + d) * (long)S_ + s];
  int dp = d ^ 64;
  float pv = qf[(h * D_ + dp) * (long)S_ + s];
  float rot = (d < 64) ? -pv : pv;
  float c = cosp[s * D_ + d], sn = sinp[s * D_ + d];
  outb[idx] = f2bf(v * c + rot * sn);
}

// ---------------------------------------------------------------------------
// Causal flash attention with WMMA; async double-buffered K/V staging.
// Grid: (S/64 query blocks, H heads); block = 128 threads = 4 waves.
// Qb [H][S][D] bf16, Kb [KV][S][D] bf16, Vt [KV][D][S] bf16,
// Ob [S][H*D] bf16  (B^T of the output projection).
// ---------------------------------------------------------------------------
#define FQ 64
#define FK 32
#define LQK 136  // 128 + 8 pad
#define LVS 40   // 32 + 8 pad

__global__ __launch_bounds__(128)
void flash_attn_kernel(const unsigned short* __restrict__ Qb,
                       const unsigned short* __restrict__ Kb,
                       const unsigned short* __restrict__ Vt,
                       unsigned short* __restrict__ Ob) {
  __shared__ __attribute__((aligned(16))) unsigned short Qs[FQ * LQK];
  __shared__ __attribute__((aligned(16))) unsigned short Ks[2][FK * LQK];
  __shared__ __attribute__((aligned(16))) unsigned short Vs[2][D_ * LVS];
  __shared__ __attribute__((aligned(16))) unsigned short Ps[4 * 16 * LVS];

  const int tid = threadIdx.x;
  const int wave = tid >> 5, lane = tid & 31;
  const int half = lane >> 4, ln = lane & 15;
  const int q0 = blockIdx.x * FQ;
  const int h = blockIdx.y;
  const int kv = h / G_;

  // stage the 64x128 Q block once (async DMA to LDS)
#pragma unroll
  for (int i = 0; i < 8; ++i) {
    int idx = tid + i * 128;
    int r = idx >> 4, c = (idx & 15) << 3;
    async_copy16(&Qs[r * LQK + c],
                 Qb + ((size_t)h * S_ + q0 + r) * D_ + c);
  }
  // K block kb=0: 32 x 128 (l, d); V^T block: 128 x 32 (d, l)
#pragma unroll
  for (int i = 0; i < 4; ++i) {
    int idx = tid + i * 128;
    int rk = idx >> 4, ck = (idx & 15) << 3;
    async_copy16(&Ks[0][rk * LQK + ck],
                 Kb + ((size_t)kv * S_ + rk) * D_ + ck);
    int dv = idx >> 2, cv = (idx & 3) << 3;
    async_copy16(&Vs[0][dv * LVS + cv],
                 Vt + ((size_t)kv * D_ + dv) * S_ + cv);
  }

  const v8f vzero = {0, 0, 0, 0, 0, 0, 0, 0};
  v8f o[8];
#pragma unroll
  for (int i = 0; i < 8; ++i) o[i] = vzero;
  float mrow[8], lrow[8];
#pragma unroll
  for (int r = 0; r < 8; ++r) { mrow[r] = -1e30f; lrow[r] = 0.f; }

  const float sc = 0.08838834764831845f;  // 1/sqrt(128)
  const int nkb = q0 / FK + 2;            // causal: only l <= q0+63
  wait_async0();
  __syncthreads();

  for (int kb = 0; kb < nkb; ++kb) {
    const int cur = kb & 1;
    const int l0 = kb * FK;
    if (kb + 1 < nkb) {  // issue next K/V block DMA before computing
      const int nxt = cur ^ 1;
      const int ln0 = (kb + 1) * FK;
#pragma unroll
      for (int i = 0; i < 4; ++i) {
        int idx = tid + i * 128;
        int rk = idx >> 4, ck = (idx & 15) << 3;
        async_copy16(&Ks[nxt][rk * LQK + ck],
                     Kb + ((size_t)kv * S_ + ln0 + rk) * D_ + ck);
        int dv = idx >> 2, cv = (idx & 3) << 3;
        async_copy16(&Vs[nxt][dv * LVS + cv],
                     Vt + ((size_t)kv * D_ + dv) * S_ + ln0 + cv);
      }
    }

    // S = Q * K^T over d: 2 key tiles, 4 k-steps of 32
    v8f st0 = vzero, st1 = vzero;
#pragma unroll
    for (int kd = 0; kd < 4; ++kd) {
      v16bf aq = load_afrag(Qs, wave * 16 + ln, LQK, kd * 32, half);
      v16bf bk0 = load_bfrag(Ks[cur], ln, LQK, kd * 32, half);
      v16bf bk1 = load_bfrag(Ks[cur], 16 + ln, LQK, kd * 32, half);
      st0 = wmma_bf16(aq, bk0, st0);
      st1 = wmma_bf16(aq, bk1, st1);
    }

    // online softmax per C-row (row = r + 8*half)
    unsigned short* pw = Ps + wave * 16 * LVS;
#pragma unroll
    for (int r = 0; r < 8; ++r) {
      int qi = q0 + wave * 16 + r + 8 * half;
      float s0 = st0[r] * sc + ((l0 + ln) > qi ? -1e9f : 0.f);
      float s1 = st1[r] * sc + ((l0 + 16 + ln) > qi ? -1e9f : 0.f);
      float bm = rmax16(fmaxf(s0, s1));
      float mn = fmaxf(mrow[r], bm);
      float esc = __expf(mrow[r] - mn);
      float p0 = __expf(s0 - mn);
      float p1 = __expf(s1 - mn);
      lrow[r] = lrow[r] * esc + rsum16(p0 + p1);
      mrow[r] = mn;
#pragma unroll
      for (int dt = 0; dt < 8; ++dt) o[dt][r] *= esc;
      // bounce P through per-wave LDS: C layout -> A-fragment layout
      pw[(r + 8 * half) * LVS + ln] = f2bf(p0);
      pw[(r + 8 * half) * LVS + 16 + ln] = f2bf(p1);
    }

    // O += P(16x32) * V(32x128): B^T = V^T rows (d, l)
    {
      v16bf ap = load_afrag(pw, ln, LVS, 0, half);
#pragma unroll
      for (int dt = 0; dt < 8; ++dt) {
        v16bf bv = load_bfrag(Vs[cur], dt * 16 + ln, LVS, 0, half);
        o[dt] = wmma_bf16(ap, bv, o[dt]);
      }
    }
    wait_async0();
    __syncthreads();
  }

  // finalize: divide by row sum, emit Ob[s][h*D + d] bf16
#pragma unroll
  for (int r = 0; r < 8; ++r) {
    float inv = 1.0f / lrow[r];
    int qi = q0 + wave * 16 + r + 8 * half;
#pragma unroll
    for (int dt = 0; dt < 8; ++dt)
      Ob[(size_t)qi * HID_ + h * D_ + dt * 16 + ln] = f2bf(o[dt][r] * inv);
  }
}

// ---------------------------------------------------------------------------
// Host-side orchestration
// ---------------------------------------------------------------------------
extern "C" void kernel_launch(void* const* d_in, const int* in_sizes, int n_in,
                              void* d_out, int out_size, void* d_ws,
                              size_t ws_size, hipStream_t stream) {
  const float* hs = (const float*)d_in[0];   // [HID][S]
  const float* wq = (const float*)d_in[1];
  const float* bq = (const float*)d_in[2];
  const float* wk = (const float*)d_in[3];
  const float* bk = (const float*)d_in[4];
  const float* wv = (const float*)d_in[5];
  const float* bv = (const float*)d_in[6];
  const float* wo = (const float*)d_in[7];
  const float* cosp = (const float*)d_in[8];  // [S][D]
  const float* sinp = (const float*)d_in[9];
  float* y = (float*)d_out;                   // [HID][S]

  char* ws = (char*)d_ws;
  size_t off = 0;
  auto carve = [&](size_t bytes) {
    size_t r = off;
    off += (bytes + 255) & ~(size_t)255;
    return r;
  };
  unsigned short* wqb = (unsigned short*)(ws + carve((size_t)HID_ * HID_ * 2));
  unsigned short* wkb = (unsigned short*)(ws + carve((size_t)KV_ * D_ * HID_ * 2));
  unsigned short* wvb = (unsigned short*)(ws + carve((size_t)KV_ * D_ * HID_ * 2));
  unsigned short* wob = (unsigned short*)(ws + carve((size_t)HID_ * HID_ * 2));
  unsigned short* xT  = (unsigned short*)(ws + carve((size_t)S_ * HID_ * 2));
  float* qf = (float*)(ws + carve((size_t)HID_ * S_ * 4));
  float* kf = (float*)(ws + carve((size_t)KV_ * D_ * S_ * 4));
  float* vf = (float*)(ws + carve((size_t)KV_ * D_ * S_ * 4));
  unsigned short* Qb = (unsigned short*)(ws + carve((size_t)H_ * S_ * D_ * 2));
  unsigned short* Kb = (unsigned short*)(ws + carve((size_t)KV_ * S_ * D_ * 2));
  unsigned short* Vt = (unsigned short*)(ws + carve((size_t)KV_ * D_ * S_ * 2));
  // attention output reuses qf's slot (qf is dead after RoPE): 16MB <= 32MB
  unsigned short* Ob = (unsigned short*)qf;

  // 1) weight casts + activation transpose-cast
  cast_bf16_kernel<<<8192, 256, 0, stream>>>(wq, wqb, (long)HID_ * HID_);
  cast_bf16_kernel<<<4096, 256, 0, stream>>>(wk, wkb, (long)KV_ * D_ * HID_);
  cast_bf16_kernel<<<4096, 256, 0, stream>>>(wv, wvb, (long)KV_ * D_ * HID_);
  cast_bf16_kernel<<<8192, 256, 0, stream>>>(wo, wob, (long)HID_ * HID_);
  transpose_cast_kernel<<<dim3(S_ / 32, HID_ / 32), dim3(32, 8), 0, stream>>>(hs, xT);

  // 2) QKV projections (WMMA)
  gemm_bf16_kernel<<<dim3(S_ / BN, HID_ / BM), 256, 0, stream>>>(
      wqb, xT, bq, qf, HID_, S_, HID_);
  gemm_bf16_kernel<<<dim3(S_ / BN, (KV_ * D_) / BM), 256, 0, stream>>>(
      wkb, xT, bk, kf, KV_ * D_, S_, HID_);
  gemm_bf16_kernel<<<dim3(S_ / BN, (KV_ * D_) / BM), 256, 0, stream>>>(
      wvb, xT, bv, vf, KV_ * D_, S_, HID_);

  // 3) RoPE + layout; V transpose-cast (layout identity: [kv*D+d][s])
  rope_kernel<<<(H_ * S_ * D_) / 256, 256, 0, stream>>>(
      qf, cosp, sinp, Qb, (long)H_ * S_ * D_);
  rope_kernel<<<(KV_ * S_ * D_) / 256, 256, 0, stream>>>(
      kf, cosp, sinp, Kb, (long)KV_ * S_ * D_);
  cast_bf16_kernel<<<4096, 256, 0, stream>>>(vf, Vt, (long)KV_ * D_ * S_);

  // 4) causal flash attention (WMMA + async K/V pipeline)
  flash_attn_kernel<<<dim3(S_ / FQ, H_), 128, 0, stream>>>(Qb, Kb, Vt, Ob);

  // 5) output projection straight into d_out (WMMA)
  gemm_bf16_kernel<<<dim3(S_ / BN, HID_ / BM), 256, 0, stream>>>(
      wob, Ob, nullptr, y, HID_, S_, HID_);
}
